// EarlyJoinGConv_13228499272260
// MI455X (gfx1250) — matrix operand order repaired and checked by reading, not compiled, not run on hardware
//
#include <hip/hip_runtime.h>

#define DEVINL __device__ __forceinline__

typedef __bf16 bf16_t;
typedef __attribute__((ext_vector_type(16))) __bf16 v16bf;
typedef __attribute__((ext_vector_type(8)))  __bf16 v8bf;
typedef __attribute__((ext_vector_type(8)))  float   v8f;

namespace {
constexpr int N_NODES = 5000;
constexpr int NP      = 5008;        // nodes padded to multiple of 16
constexpr int C       = 32;
constexpr int H       = 64;
constexpr int NC      = N_NODES * C; // 160000 rows of the (n,c) flattened tensor
constexpr int K0      = 160;         // layer-0 node-feature K (148 padded to 160)
}

union ABf { struct { v8bf lo, hi; } h; v16bf v; };

DEVINL v8f bwmma(v16bf a, v16bf b, v8f c) {
  return __builtin_amdgcn_wmma_f32_16x16x32_bf16(false, a, false, b, (short)0, c,
                                                 false, false);
}

// A fragment (16x32 bf16, row-major source). rowp = X + (R + (lane&15))*ld + kbase.
// Lane<16 holds K[0..7] and K[16..23]; lane>=16 holds K[8..15] and K[24..31].
DEVINL v16bf ld_afrag(const bf16_t* rowp, int lane) {
  const bf16_t* p = rowp + ((lane >> 4) << 3);
  ABf u;
  u.h.lo = *(const v8bf*)(p);
  u.h.hi = *(const v8bf*)(p + 16);
  return u.v;
}

// B fragment from pre-packed weights: 32 contiguous bytes per lane per fragment.
DEVINL v16bf ld_bfrag(const bf16_t* pack, int frag, int lane) {
  const bf16_t* p = pack + ((size_t)frag << 9) + (lane << 4);
  ABf u;
  u.h.lo = *(const v8bf*)(p);
  u.h.hi = *(const v8bf*)(p + 8);
  return u.v;
}

// ---- pack a (K x 64) f32 weight (row-major, row stride 64) into B fragments ----
// frag f = kstep*4 + jtile ; lane L group g=L>>4 holds K = kstep*32 + 8*g + (e&7) + 16*(e>>3),
// column = jtile*16 + (L&15). Zero-pad K beyond valid range.
__global__ void pack_kernel(const float* __restrict__ w, bf16_t* __restrict__ dst,
                            int K, int ksteps) {
  int t = blockIdx.x * blockDim.x + threadIdx.x;
  if (t >= ksteps * 2048) return;
  int e = t & 15;
  int L = (t >> 4) & 31;
  int f = t >> 9;
  int kstep = f >> 2, j = f & 3;
  int k   = kstep * 32 + ((L >> 4) << 3) + (e & 7) + ((e >> 3) << 4);
  int col = (j << 4) + (L & 15);
  float v = (k < K) ? w[k * 64 + col] : 0.f;
  dst[((size_t)f << 9) + (L << 4) + e] = (bf16_t)v;
}

// ---- xn = [node_feat | op_emb[opcode]] padded to 160 cols, bf16 ----
__global__ void build_xn_kernel(const float* __restrict__ nf, const int* __restrict__ opc,
                                const float* __restrict__ opemb, bf16_t* __restrict__ xn) {
  int i = blockIdx.x * blockDim.x + threadIdx.x;
  if (i >= NP * K0) return;
  int k = i % K0;
  int n = i / K0;
  float v = 0.f;
  if (n < N_NODES) {
    if (k < 140)      v = nf[n * 140 + k];
    else if (k < 148) v = opemb[opc[n] * 8 + (k - 140)];
  }
  xn[i] = (bf16_t)v;
}

// ---- bidirectional degree ----
__global__ void deg_kernel(const int* __restrict__ ei, float* __restrict__ deg, int m) {
  int e = blockIdx.x * blockDim.x + threadIdx.x;
  if (e >= 2 * m) return;
  int dst = (e < m) ? ei[2 * e + 1] : ei[2 * (e - m) + 0];
  atomicAdd(&deg[dst], 1.0f);
}

// ---- bidirectional scatter-add of bf16 rows into f32 accumulator ----
__global__ void scatter_kernel(const bf16_t* __restrict__ x, const int* __restrict__ ei,
                               float* __restrict__ aggf, int m, int L) {
  int idx = blockIdx.x * blockDim.x + threadIdx.x;
  int quads = L >> 2;
  int total = 2 * m * quads;
  if (idx >= total) return;
  int q = idx % quads;
  int e = idx / quads;
  int src, dst;
  if (e < m) { src = ei[2 * e];          dst = ei[2 * e + 1]; }
  else       { src = ei[2 * (e - m) + 1]; dst = ei[2 * (e - m)]; }
  const bf16_t* xp = x + (size_t)src * L + q * 4;
  float* ap = aggf + (size_t)dst * L + q * 4;
  atomicAdd(ap + 0, (float)xp[0]);
  atomicAdd(ap + 1, (float)xp[1]);
  atomicAdd(ap + 2, (float)xp[2]);
  atomicAdd(ap + 3, (float)xp[3]);
}

// ---- agg_bf16 = aggf / max(deg,1) ----
__global__ void finalize_kernel(const float* __restrict__ aggf, const float* __restrict__ deg,
                                bf16_t* __restrict__ aggbf, int L, int total) {
  int i = blockIdx.x * blockDim.x + threadIdx.x;
  if (i >= total) return;
  int n = i / L;
  float inv = 1.0f / fmaxf(deg[n], 1.0f);
  aggbf[i] = (bf16_t)(aggf[i] * inv);
}

// ---- layer-0 node-part GEMMs: S0 = xn@Ws0, N0 = aggxn@Wn0 (K=160, f32 out) ----
DEVINL void gemm160_tile(const bf16_t* __restrict__ X, const bf16_t* __restrict__ WP,
                         float* __restrict__ O, int R, int lane) {
  const bf16_t* rowp = X + (size_t)(R + (lane & 15)) * K0;
  v16bf a[5];
#pragma unroll
  for (int s = 0; s < 5; ++s) a[s] = ld_afrag(rowp + s * 32, lane);
  int mr = (lane >> 4) << 3;
  int cl = lane & 15;
#pragma unroll
  for (int j = 0; j < 4; ++j) {
    v8f acc = {};
#pragma unroll
    for (int s = 0; s < 5; ++s) acc = bwmma(a[s], ld_bfrag(WP, s * 4 + j, lane), acc);
    int col = (j << 4) + cl;
#pragma unroll
    for (int v = 0; v < 8; ++v) O[(size_t)(R + mr + v) * 64 + col] = acc[v];
  }
}

__global__ void __launch_bounds__(128)
gemm160_kernel(const bf16_t* __restrict__ Xa, const bf16_t* __restrict__ Xb,
               const bf16_t* __restrict__ WaP, const bf16_t* __restrict__ WbP,
               float* __restrict__ Oa, float* __restrict__ Ob, int ntiles) {
  int lane = threadIdx.x & 31;
  int wg = blockIdx.x * (blockDim.x >> 5) + (threadIdx.x >> 5);
  int nw = gridDim.x * (blockDim.x >> 5);
  for (int t = wg; t < ntiles; t += nw) {
    int R = t << 4;
    gemm160_tile(Xa, WaP, Oa, R, lane);
    gemm160_tile(Xb, WbP, Ob, R, lane);
  }
}

// ---- config tails: Sc = cf@Ws0[148:], Nc = cf@Wn0[148:]  (tiny) ----
__global__ void cfg_kernel(const float* __restrict__ cf, const float* __restrict__ ws0,
                           const float* __restrict__ wn0, float* __restrict__ Sc,
                           float* __restrict__ Nc) {
  int t = blockIdx.x * blockDim.x + threadIdx.x;
  if (t >= 4096) return;
  int which = t >> 11;
  int c = (t >> 6) & 31;
  int h = t & 63;
  const float* w = which ? wn0 : ws0;
  float s = 0.f;
  for (int k = 0; k < 24; ++k) s += cf[c * 24 + k] * w[(148 + k) * 64 + h];
  (which ? Nc : Sc)[c * 64 + h] = s;
}

// ---- x1[n,c,h] = relu(S0 + Sc + N0 + (deg>0 ? Nc : 0) + b0) ----
__global__ void compose_x1_kernel(const float* __restrict__ S0, const float* __restrict__ N0,
                                  const float* __restrict__ Sc, const float* __restrict__ Nc,
                                  const float* __restrict__ b0, const float* __restrict__ deg,
                                  bf16_t* __restrict__ x1, int total) {
  int i = blockIdx.x * blockDim.x + threadIdx.x;
  if (i >= total) return;
  int h = i & 63;
  int c = (i >> 6) & 31;
  int n = i >> 11;
  float z = S0[n * 64 + h] + N0[n * 64 + h] + Sc[c * 64 + h] + b0[h];
  if (deg[n] > 0.f) z += Nc[c * 64 + h];
  x1[i] = (bf16_t)fmaxf(z, 0.f);
}

// ---- fused SAGE + skip-pool: Xout = relu(X@Ws + A@Wn + b);
//      pooled[c, off+h] += sum_n relu(X@Wsk + sb) ----
__global__ void __launch_bounds__(128)
sage_skip_kernel(const bf16_t* __restrict__ Xin, const bf16_t* __restrict__ Agg,
                 bf16_t* __restrict__ Xout,
                 const bf16_t* __restrict__ WsP, const bf16_t* __restrict__ WnP,
                 const bf16_t* __restrict__ WkP,
                 const float* __restrict__ bias, const float* __restrict__ skip_bias,
                 float* __restrict__ pooled, int pool_off, int ntiles, int do_sage) {
  __shared__ float lpool[C * H];
  int tid = threadIdx.x;
  for (int i = tid; i < C * H; i += blockDim.x) lpool[i] = 0.f;
  __syncthreads();

  int lane = tid & 31;
  int wg = blockIdx.x * (blockDim.x >> 5) + (tid >> 5);
  int nw = gridDim.x * (blockDim.x >> 5);

  for (int t = wg; t < ntiles; t += nw) {
    int R = t << 4;
    int tn = t + nw;
    if (tn < ntiles) {
      __builtin_prefetch(Xin + ((size_t)tn << 4) * 64, 0, 1);
      if (do_sage) __builtin_prefetch(Agg + ((size_t)tn << 4) * 64, 0, 1);
    }
    const bf16_t* xrow = Xin + (size_t)(R + (lane & 15)) * 64;
    v16bf ax0 = ld_afrag(xrow, lane);
    v16bf ax1 = ld_afrag(xrow + 32, lane);
    int cl = lane & 15;
    int hgrp = (lane >> 4) << 3;

    // skip head on Xin -> LDS pooled accumulator
#pragma unroll
    for (int j = 0; j < 4; ++j) {
      v8f acc = {};
      acc = bwmma(ax0, ld_bfrag(WkP, j, lane), acc);
      acc = bwmma(ax1, ld_bfrag(WkP, 4 + j, lane), acc);
      int col = (j << 4) + cl;
      float sb = skip_bias[col];
      int cbase = (R & 31) + hgrp;  // config index of D row M=0 for this lane half
#pragma unroll
      for (int v = 0; v < 8; ++v) {
        float z = fmaxf(acc[v] + sb, 0.f);
        atomicAdd(&lpool[(cbase + v) * H + col], z);
      }
    }

    if (do_sage) {
      const bf16_t* arow = Agg + (size_t)(R + (lane & 15)) * 64;
      v16bf aa0 = ld_afrag(arow, lane);
      v16bf aa1 = ld_afrag(arow + 32, lane);
#pragma unroll
      for (int j = 0; j < 4; ++j) {
        v8f acc = {};
        acc = bwmma(ax0, ld_bfrag(WsP, j, lane), acc);
        acc = bwmma(ax1, ld_bfrag(WsP, 4 + j, lane), acc);
        acc = bwmma(aa0, ld_bfrag(WnP, j, lane), acc);
        acc = bwmma(aa1, ld_bfrag(WnP, 4 + j, lane), acc);
        int col = (j << 4) + cl;
        float bb = bias[col];
#pragma unroll
        for (int v = 0; v < 8; ++v) {
          float z = fmaxf(acc[v] + bb, 0.f);
          Xout[(size_t)(R + hgrp + v) * 64 + col] = (bf16_t)z;
        }
      }
    }
  }
  __syncthreads();
  for (int i = tid; i < C * H; i += blockDim.x)
    atomicAdd(&pooled[(i >> 6) * 192 + pool_off + (i & 63)], lpool[i]);
}

// ---- final MLP: (32,192) -> relu 128 -> relu 64 -> 1 ----
__global__ void mlp_kernel(const float* __restrict__ pooled,
                           const float* __restrict__ p1w, const float* __restrict__ p1b,
                           const float* __restrict__ p2w, const float* __restrict__ p2b,
                           const float* __restrict__ p3w, const float* __restrict__ p3b,
                           float* __restrict__ out) {
  __shared__ float h1[32 * 128];
  __shared__ float h2[32 * 64];
  int tid = threadIdx.x;
  for (int i = tid; i < 32 * 128; i += blockDim.x) {
    int c = i >> 7, j = i & 127;
    float s = p1b[j];
    for (int k = 0; k < 192; ++k) s += pooled[c * 192 + k] * p1w[k * 128 + j];
    h1[i] = fmaxf(s, 0.f);
  }
  __syncthreads();
  for (int i = tid; i < 32 * 64; i += blockDim.x) {
    int c = i >> 6, j = i & 63;
    float s = p2b[j];
    for (int k = 0; k < 128; ++k) s += h1[c * 128 + k] * p2w[k * 64 + j];
    h2[i] = fmaxf(s, 0.f);
  }
  __syncthreads();
  if (tid < 32) {
    float s = p3b[0];
    for (int k = 0; k < 64; ++k) s += h2[tid * 64 + k] * p3w[k];
    out[tid] = s;
  }
}

extern "C" void kernel_launch(void* const* d_in, const int* in_sizes, int n_in,
                              void* d_out, int out_size, void* d_ws, size_t ws_size,
                              hipStream_t stream) {
  (void)n_in; (void)out_size; (void)ws_size;
  const float* nf       = (const float*)d_in[0];
  const int*   opc      = (const int*)d_in[1];
  const int*   ei       = (const int*)d_in[2];
  const float* cfg      = (const float*)d_in[3];
  const float* opemb    = (const float*)d_in[4];
  const float* w_self0  = (const float*)d_in[5];
  const float* w_neigh0 = (const float*)d_in[6];
  const float* b0       = (const float*)d_in[7];
  const float* w_self   = (const float*)d_in[8];
  const float* w_neigh  = (const float*)d_in[9];
  const float* bl       = (const float*)d_in[10];
  const float* skip_w   = (const float*)d_in[11];
  const float* skip_b   = (const float*)d_in[12];
  const float* p1w      = (const float*)d_in[13];
  const float* p1b      = (const float*)d_in[14];
  const float* p2w      = (const float*)d_in[15];
  const float* p2b      = (const float*)d_in[16];
  const float* p3w      = (const float*)d_in[17];
  const float* p3b      = (const float*)d_in[18];
  float* out = (float*)d_out;
  int m = in_sizes[2] / 2;  // 8000 undirected edges

  char* wsb = (char*)d_ws;
  size_t off = 0;
  auto carve = [&](size_t bytes) -> void* {
    void* p = (void*)(wsb + off);
    off += (bytes + 255) & ~(size_t)255;
    return p;
  };
  float*  deg     = (float*)carve((size_t)NP * 4);
  bf16_t* xnbf    = (bf16_t*)carve((size_t)NP * K0 * 2);
  float*  aggxnf  = (float*)carve((size_t)NP * K0 * 4);
  bf16_t* aggxnbf = (bf16_t*)carve((size_t)NP * K0 * 2);
  float*  S0      = (float*)carve((size_t)NP * 64 * 4);
  float*  N0      = (float*)carve((size_t)NP * 64 * 4);
  float*  Sc      = (float*)carve(32 * 64 * 4);
  float*  Nc      = (float*)carve(32 * 64 * 4);
  bf16_t* x1      = (bf16_t*)carve((size_t)NC * 64 * 2);  // also reused as x3
  bf16_t* x2      = (bf16_t*)carve((size_t)NC * 64 * 2);
  float*  aggf    = (float*)carve((size_t)NC * 64 * 4);
  bf16_t* aggbf   = (bf16_t*)carve((size_t)NC * 64 * 2);
  float*  pooled  = (float*)carve(32 * 192 * 4);
  bf16_t* ws0p    = (bf16_t*)carve(20 * 512 * 2);
  bf16_t* wn0p    = (bf16_t*)carve(20 * 512 * 2);
  bf16_t* ws1p    = (bf16_t*)carve(8 * 512 * 2);
  bf16_t* ws2p    = (bf16_t*)carve(8 * 512 * 2);
  bf16_t* wn1p    = (bf16_t*)carve(8 * 512 * 2);
  bf16_t* wn2p    = (bf16_t*)carve(8 * 512 * 2);
  bf16_t* sk0p    = (bf16_t*)carve(8 * 512 * 2);
  bf16_t* sk1p    = (bf16_t*)carve(8 * 512 * 2);
  bf16_t* sk2p    = (bf16_t*)carve(8 * 512 * 2);

  // ---- weight packing (tiny, cache-resident) ----
  pack_kernel<<<(5 * 2048 + 255) / 256, 256, 0, stream>>>(w_self0, ws0p, 148, 5);
  pack_kernel<<<(5 * 2048 + 255) / 256, 256, 0, stream>>>(w_neigh0, wn0p, 148, 5);
  pack_kernel<<<(2 * 2048 + 255) / 256, 256, 0, stream>>>(w_self, ws1p, 64, 2);
  pack_kernel<<<(2 * 2048 + 255) / 256, 256, 0, stream>>>(w_self + 4096, ws2p, 64, 2);
  pack_kernel<<<(2 * 2048 + 255) / 256, 256, 0, stream>>>(w_neigh, wn1p, 64, 2);
  pack_kernel<<<(2 * 2048 + 255) / 256, 256, 0, stream>>>(w_neigh + 4096, wn2p, 64, 2);
  pack_kernel<<<(2 * 2048 + 255) / 256, 256, 0, stream>>>(skip_w, sk0p, 64, 2);
  pack_kernel<<<(2 * 2048 + 255) / 256, 256, 0, stream>>>(skip_w + 4096, sk1p, 64, 2);
  pack_kernel<<<(2 * 2048 + 255) / 256, 256, 0, stream>>>(skip_w + 8192, sk2p, 64, 2);

  // ---- layer 0 via rank factorization ----
  hipMemsetAsync(deg, 0, (size_t)NP * 4, stream);
  hipMemsetAsync(aggxnf, 0, (size_t)NP * K0 * 4, stream);
  hipMemsetAsync(pooled, 0, 32 * 192 * 4, stream);
  build_xn_kernel<<<(NP * K0 + 255) / 256, 256, 0, stream>>>(nf, opc, opemb, xnbf);
  deg_kernel<<<(2 * m + 255) / 256, 256, 0, stream>>>(ei, deg, m);
  scatter_kernel<<<(2 * m * (K0 / 4) + 255) / 256, 256, 0, stream>>>(xnbf, ei, aggxnf, m, K0);
  finalize_kernel<<<(NP * K0 + 255) / 256, 256, 0, stream>>>(aggxnf, deg, aggxnbf, K0, NP * K0);
  gemm160_kernel<<<40, 128, 0, stream>>>(xnbf, aggxnbf, ws0p, wn0p, S0, N0, NP / 16);
  cfg_kernel<<<16, 256, 0, stream>>>(cfg, w_self0, w_neigh0, Sc, Nc);
  compose_x1_kernel<<<(NC * 64 + 255) / 256, 256, 0, stream>>>(S0, N0, Sc, Nc, b0, deg, x1,
                                                               NC * 64);

  // ---- layer 1: x2 = sage(x1); pool skip0(x1) ----
  hipMemsetAsync(aggf, 0, (size_t)NC * 64 * 4, stream);
  scatter_kernel<<<(2 * m * 512 + 255) / 256, 256, 0, stream>>>(x1, ei, aggf, m, 2048);
  finalize_kernel<<<(NC * 64 + 255) / 256, 256, 0, stream>>>(aggf, deg, aggbf, 2048, NC * 64);
  sage_skip_kernel<<<256, 128, 0, stream>>>(x1, aggbf, x2, ws1p, wn1p, sk0p,
                                            bl, skip_b, pooled, 0, NC / 16, 1);

  // ---- layer 2: x3 (reusing x1) = sage(x2); pool skip1(x2) ----
  hipMemsetAsync(aggf, 0, (size_t)NC * 64 * 4, stream);
  scatter_kernel<<<(2 * m * 512 + 255) / 256, 256, 0, stream>>>(x2, ei, aggf, m, 2048);
  finalize_kernel<<<(NC * 64 + 255) / 256, 256, 0, stream>>>(aggf, deg, aggbf, 2048, NC * 64);
  sage_skip_kernel<<<256, 128, 0, stream>>>(x2, aggbf, x1, ws2p, wn2p, sk1p,
                                            bl + 64, skip_b + 64, pooled, 64, NC / 16, 1);

  // ---- skip2(x3) only ----
  sage_skip_kernel<<<256, 128, 0, stream>>>(x1, nullptr, nullptr, nullptr, nullptr, sk2p,
                                            nullptr, skip_b + 128, pooled, 128, NC / 16, 0);

  // ---- final MLP ----
  mlp_kernel<<<1, 256, 0, stream>>>(pooled, p1w, p1b, p2w, p2b, p3w, p3b, out);
}
